// STARLIVTSModel_75952201662655
// MI455X (gfx1250) — matrix operand order repaired and checked by reading, not compiled, not run on hardware
//
#include <hip/hip_runtime.h>
#include <hip/hip_bf16.h>
#include <math.h>

// Problem constants
#define BB 32
#define LL 4096
#define CC 64
#define DD 512
#define HH 720
#define EPS 1e-5f

typedef _Float16 half16 __attribute__((ext_vector_type(16)));
typedef _Float16 half8  __attribute__((ext_vector_type(8)));
typedef float    float8 __attribute__((ext_vector_type(8)));
typedef unsigned uint4v __attribute__((ext_vector_type(4)));
typedef unsigned uint8v __attribute__((ext_vector_type(8)));

static __device__ __forceinline__ float8 wmma_f16(half16 a, half16 b, float8 c) {
    // D = A(16x32 f16) * B(32x16 f16) + C(16x16 f32)
    return __builtin_amdgcn_wmma_f32_16x16x32_f16(
        false, a, false, b, (short)0, c, false, false);
}

#define SHUF16(lo, hi) __builtin_shufflevector((lo), (hi), 0,1,2,3,4,5,6,7,8,9,10,11,12,13,14,15)

// ---------------------------------------------------------------------------
// Fragment-ready packing (ISA 7.12.2, wave32, 16-bit operands):
//   A frag (16x32): lane m = lane&15, holds K = ((lane<16)?0:16)+e  (contiguous, this file's order)
//   B frag (32x16): lane n = lane&15, holds K = kb+e (e<8), kb+8+e (e>=8),
//                   kb = (lane<16)?0:8   (two contiguous 8-half runs)
// Packed storage: dst[((tile*nKc + kc)*32 + lane)*16 + e]  -> one 32B load/lane.
// ---------------------------------------------------------------------------
__global__ void pack_b_kernel(const float* __restrict__ W, _Float16* __restrict__ dst,
                              int Ntot, int nKc) {
    const int nt = blockIdx.x, kc = blockIdx.y, lane = threadIdx.x;
    const int kb = (lane < 16) ? 0 : 8;
    const int n  = nt * 16 + (lane & 15);
    const size_t base = (((size_t)nt * nKc + kc) * 32 + lane) * 16;
    #pragma unroll
    for (int e = 0; e < 16; ++e)
        dst[base + e] = (_Float16)W[(size_t)(kc * 32 + kb + e + ((e < 8) ? 0 : 8)) * Ntot + n];
}

__global__ void pack_a_kernel(const float* __restrict__ tw, _Float16* __restrict__ dst) {
    const int mt = blockIdx.x, kc = blockIdx.y, lane = threadIdx.x;
    const int kb = (lane < 16) ? 0 : 16;
    const int h  = mt * 16 + (lane & 15);
    const size_t base = (((size_t)mt * 128 + kc) * 32 + lane) * 16;
    #pragma unroll
    for (int e = 0; e < 16; ++e) {
        const int kk = kb + e;
        dst[base + e] = (_Float16)tw[(size_t)(kc * 32 + kk) * HH + h];
    }
}

// ---------------------------------------------------------------------------
// Kernel 1: per-(b,c) mean / stdev over L   (<<<B, C>>>)
// ---------------------------------------------------------------------------
__global__ void stats_kernel(const float* __restrict__ x,
                             float* __restrict__ mean, float* __restrict__ stdv) {
    const int b = blockIdx.x;
    const int c = threadIdx.x;
    const float* xp = x + (size_t)b * LL * CC + c;
    float s = 0.f, s2 = 0.f;
    for (int l = 0; l < LL; ++l) {
        float v = xp[(size_t)l * CC];
        s += v; s2 += v * v;
    }
    const float m = s * (1.0f / LL);
    float var = s2 * (1.0f / LL) - m * m;
    var = fmaxf(var, 0.0f);
    mean[b * CC + c] = m;
    stdv[b * CC + c] = sqrtf(var + EPS);
}

// ---------------------------------------------------------------------------
// Kernel 2: RevIN normalize + EMA scan -> seasonal/trend (f16)  (<<<B, C>>>)
// ---------------------------------------------------------------------------
__global__ void scan_kernel(const float* __restrict__ x,
                            const float* __restrict__ alpha,
                            const float* __restrict__ rev_w,
                            const float* __restrict__ rev_b,
                            const float* __restrict__ mean,
                            const float* __restrict__ stdv,
                            _Float16* __restrict__ seas,
                            _Float16* __restrict__ tren) {
    const int b = blockIdx.x;
    const int c = threadIdx.x;
    const float a  = 1.0f / (1.0f + expf(-alpha[c]));
    const float w  = rev_w[c];
    const float bb = rev_b[c];
    const float m  = mean[b * CC + c];
    const float rs = 1.0f / stdv[b * CC + c];
    const size_t base = (size_t)b * LL * CC + c;
    float carry = 0.0f;
    for (int l = 0; l < LL; ++l) {
        const float xv = x[base + (size_t)l * CC];
        const float xn = (xv - m) * rs * w + bb;
        const float tr = (l == 0) ? xn : (a * carry + (1.0f - a) * xn);
        carry = tr;
        seas[base + (size_t)l * CC] = (_Float16)(xn - tr);
        tren[base + (size_t)l * CC] = (_Float16)tr;
    }
}

// ---------------------------------------------------------------------------
// Kernel 3: fused per-row MLP. One wave per 16-row tile; 2 waves/block.
//  GEMM1: z(16x64) @ ew(64x512) + eb -> LDS (f16)
//  GEMM2: LDS(16x512) @ pw(512x64) + pb -> gpack (B-fragment-packed)
// ---------------------------------------------------------------------------
__global__ void mlp_kernel(const _Float16* __restrict__ z,
                           const _Float16* __restrict__ ewp,
                           const float*    __restrict__ eb,
                           const _Float16* __restrict__ pwp,
                           const float*    __restrict__ pb,
                           _Float16* __restrict__ gpack) {
    __shared__ _Float16 hbuf[2][16][DD + 16];

    const int lane = threadIdx.x & 31;
    const int wave = threadIdx.x >> 5;
    const int rowtile = blockIdx.x * 2 + wave;
    const int b  = rowtile >> 8;
    const int rt = rowtile & 255;

    const int mrow = lane & 15;
    const int ncol = lane & 15;
    const int kb_a = (lane < 16) ? 0 : 8;
    const int hi8  = (lane >= 16) ? 8 : 0;

    const _Float16* zr = z + ((size_t)rowtile * 16 + mrow) * CC;
    half8 l0 = *(const half8*)(zr + kb_a);
    half8 h0 = *(const half8*)(zr + kb_a + 16);
    half8 l1 = *(const half8*)(zr + 32 + kb_a);
    half8 h1 = *(const half8*)(zr + 32 + kb_a + 16);
    half16 a0 = SHUF16(l0, h0);
    half16 a1 = SHUF16(l1, h1);

    for (int nt = 0; nt < 32; ++nt) {
        half16 b0 = *(const half16*)(ewp + (((size_t)nt * 2 + 0) * 32 + lane) * 16);
        half16 b1 = *(const half16*)(ewp + (((size_t)nt * 2 + 1) * 32 + lane) * 16);
        float8 acc = {};
        acc = wmma_f16(a0, b0, acc);
        acc = wmma_f16(a1, b1, acc);
        const int   n    = nt * 16 + ncol;
        const float bias = eb[n];
        #pragma unroll
        for (int r = 0; r < 8; ++r)
            hbuf[wave][r + hi8][n] = (_Float16)(acc[r] + bias);
    }

    for (int ct = 0; ct < 4; ++ct) {
        float8 acc = {};
        for (int kc = 0; kc < 16; ++kc) {
            const int k0 = kc * 32;
            half8 lo = *(const half8*)&hbuf[wave][mrow][k0 + kb_a];
            half8 hi = *(const half8*)&hbuf[wave][mrow][k0 + kb_a + 16];
            half16 af = SHUF16(lo, hi);
            half16 bf = *(const half16*)(pwp + (((size_t)ct * 16 + kc) * 32 + lane) * 16);
            acc = wmma_f16(af, bf, acc);
        }
        const float bias = pb[ct * 16 + ncol];
        half8 ov;
        #pragma unroll
        for (int r = 0; r < 8; ++r)
            ov[r] = (_Float16)(acc[r] + bias);
        const size_t dst = ((((size_t)b * 128 + (rt >> 1)) * 4 + ct) * 32
                            + (ncol + (rt & 1) * 16)) * 16 + hi8;
        *(half8*)(gpack + dst) = ov;
    }
}

// ---------------------------------------------------------------------------
// Kernel 4: temporal mix + combine + inverse RevIN.
// Grid (45, 32); 4 waves = 4 c-tiles. A fragments (tw, shared by all 4 waves)
// are double-buffered into LDS: seasonal path via the Tensor Data Mover
// (tensor_load_to_lds, TENSORcnt), trend path via async loads (ASYNCcnt).
// Per wave: M=16(h), N=16(c), K=4096 -> 256 WMMAs.
// ---------------------------------------------------------------------------
__global__ void temporal_kernel(const _Float16* __restrict__ gpack_s,
                                const _Float16* __restrict__ gpack_t,
                                const _Float16* __restrict__ ap_s,
                                const _Float16* __restrict__ ap_t,
                                const float* __restrict__ sb,
                                const float* __restrict__ tb,
                                const float* __restrict__ rev_w,
                                const float* __restrict__ rev_b,
                                const float* __restrict__ mean,
                                const float* __restrict__ stdv,
                                float* __restrict__ out) {
    __shared__ _Float16 abuf[2][2][32 * 16];   // [path][double-buffer][lane*16]

    const int lane  = threadIdx.x & 31;
    const int wave  = threadIdx.x >> 5;
    const int mtile = blockIdx.x;
    const int b     = blockIdx.y;
    const int ncol  = lane & 15;
    const int c     = wave * 16 + ncol;

    // --- TDM copy: 1KB of seasonal A fragments -> abuf[0][bufi] (wave-uniform)
    auto issue_tdm = [&](int kc, int bufi) {
        const unsigned long long ga =
            (unsigned long long)(uintptr_t)(ap_s + (((size_t)mtile * 128 + kc) * 32) * 16);
        const unsigned lds = (unsigned)(uintptr_t)&abuf[0][bufi][0];   // low 32b = LDS offset
        uint4v g0;
        g0[0] = 1u;                                   // count=1 (valid descriptor)
        g0[1] = lds;                                  // lds_addr
        g0[2] = (unsigned)ga;                         // global_addr[31:0]
        g0[3] = (unsigned)((ga >> 32) & 0x01FFFFFFu)  // global_addr[56:32]
                | 0x80000000u;                        // type=2 ("image")
        uint8v g1;
        g1[0] = 2u << 16;          // workgroup_mask=0, data_size=2 (4B elements)
        g1[1] = 256u << 16;        // tensor_dim0 = 256 dwords (low 16 bits)
        g1[2] = (0u) | (1u << 16); // tensor_dim0 hi=0, tensor_dim1 = 1
        g1[3] = (0u) | (256u << 16); // tensor_dim1 hi=0, tile_dim0 = 256
        g1[4] = 1u;                // tile_dim1 = 1, tile_dim2 = 0
        g1[5] = 256u;              // tensor_dim0_stride[31:0]
        g1[6] = 256u << 16;        // stride0 hi=0, tensor_dim1_stride[15:0]=256
        g1[7] = 0u;                // tensor_dim1_stride hi
        asm volatile("tensor_load_to_lds %0, %1" :: "s"(g0), "s"(g1) : "memory");
    };

    // --- async copy: trend A fragments -> abuf[1][bufi] (per-lane 32B)
    auto issue_async = [&](int kc, int bufi) {
        const unsigned long long gt64 = (unsigned long long)(uintptr_t)
            (ap_t + (((size_t)mtile * 128 + kc) * 32 + lane) * 16);
        const unsigned lt = (unsigned)(uintptr_t)&abuf[1][bufi][lane * 16];
        asm volatile("global_load_async_to_lds_b128 %0, %1, off"
                     :: "v"(lt), "v"(gt64) : "memory");
        asm volatile("global_load_async_to_lds_b128 %0, %1, off offset:16"
                     :: "v"(lt), "v"(gt64) : "memory");
    };

    if (wave == 0) { issue_tdm(0, 0); issue_async(0, 0); }
    __builtin_amdgcn_s_wait_tensorcnt(0);
    asm volatile("s_wait_asynccnt 0" ::: "memory");
    __syncthreads();

    float8 acc = {};
    for (int kc = 0; kc < 128; ++kc) {
        const int cb = kc & 1;
        if (wave == 0 && kc < 127) { issue_tdm(kc + 1, cb ^ 1); issue_async(kc + 1, cb ^ 1); }

        const size_t gb = ((((size_t)b * 128 + kc) * 4 + wave) * 32 + lane) * 16;
        __builtin_prefetch(gpack_s + gb + 4 * 32 * 16, 0, 1);
        __builtin_prefetch(gpack_t + gb + 4 * 32 * 16, 0, 1);

        half16 as_ = *(const half16*)&abuf[0][cb][lane * 16];  // ds_load_b128 x2
        half16 at_ = *(const half16*)&abuf[1][cb][lane * 16];
        half16 bs_ = *(const half16*)(gpack_s + gb);           // global b128 x2
        half16 bt_ = *(const half16*)(gpack_t + gb);

        acc = wmma_f16(as_, bs_, acc);
        acc = wmma_f16(at_, bt_, acc);

        __builtin_amdgcn_s_wait_tensorcnt(0);
        asm volatile("s_wait_asynccnt 0" ::: "memory");
        __syncthreads();
    }

    const float rw = rev_w[c] + EPS;
    const float rb = rev_b[c];
    const float sd = stdv[b * CC + c];
    const float mn = mean[b * CC + c];
    #pragma unroll
    for (int r = 0; r < 8; ++r) {
        const int h = mtile * 16 + r + ((lane >= 16) ? 8 : 0);
        float v = acc[r] + sb[h] + tb[h];
        v = (v - rb) / rw;
        v = v * sd + mn;
        out[((size_t)b * HH + h) * CC + c] = v;
    }
}

// ---------------------------------------------------------------------------
// Host launcher
// ---------------------------------------------------------------------------
extern "C" void kernel_launch(void* const* d_in, const int* in_sizes, int n_in,
                              void* d_out, int out_size, void* d_ws, size_t ws_size,
                              hipStream_t stream) {
    const float* x     = (const float*)d_in[0];
    const float* alpha = (const float*)d_in[1];
    const float* rev_w = (const float*)d_in[2];
    const float* rev_b = (const float*)d_in[3];
    const float* se_w  = (const float*)d_in[4];
    const float* se_b  = (const float*)d_in[5];
    const float* sp_w  = (const float*)d_in[6];
    const float* sp_b  = (const float*)d_in[7];
    const float* st_w  = (const float*)d_in[8];
    const float* st_b  = (const float*)d_in[9];
    const float* te_w  = (const float*)d_in[10];
    const float* te_b  = (const float*)d_in[11];
    const float* tp_w  = (const float*)d_in[12];
    const float* tp_b  = (const float*)d_in[13];
    const float* tt_w  = (const float*)d_in[14];
    const float* tt_b  = (const float*)d_in[15];
    float* out = (float*)d_out;

    char* ws = (char*)d_ws;
    size_t off = 0;
    auto alloc = [&](size_t bytes) -> void* {
        void* p = ws + off;
        off += (bytes + 255) & ~(size_t)255;
        return p;
    };

    const size_t NBL = (size_t)BB * LL;
    float*    mean = (float*)   alloc((size_t)BB * CC * sizeof(float));
    float*    stdv = (float*)   alloc((size_t)BB * CC * sizeof(float));
    _Float16* seas = (_Float16*)alloc(NBL * CC * sizeof(_Float16));
    _Float16* tren = (_Float16*)alloc(NBL * CC * sizeof(_Float16));
    _Float16* gp_s = (_Float16*)alloc(NBL * CC * sizeof(_Float16));
    _Float16* gp_t = (_Float16*)alloc(NBL * CC * sizeof(_Float16));
    _Float16* ewp_s = (_Float16*)alloc((size_t)CC * DD * sizeof(_Float16));
    _Float16* pwp_s = (_Float16*)alloc((size_t)DD * CC * sizeof(_Float16));
    _Float16* ewp_t = (_Float16*)alloc((size_t)CC * DD * sizeof(_Float16));
    _Float16* pwp_t = (_Float16*)alloc((size_t)DD * CC * sizeof(_Float16));
    _Float16* awp_s = (_Float16*)alloc((size_t)LL * HH * sizeof(_Float16));
    _Float16* awp_t = (_Float16*)alloc((size_t)LL * HH * sizeof(_Float16));
    (void)ws_size; (void)n_in; (void)in_sizes; (void)out_size;

    pack_b_kernel<<<dim3(DD / 16, 2),  32, 0, stream>>>(se_w, ewp_s, DD, 2);
    pack_b_kernel<<<dim3(CC / 16, 16), 32, 0, stream>>>(sp_w, pwp_s, CC, 16);
    pack_b_kernel<<<dim3(DD / 16, 2),  32, 0, stream>>>(te_w, ewp_t, DD, 2);
    pack_b_kernel<<<dim3(CC / 16, 16), 32, 0, stream>>>(tp_w, pwp_t, CC, 16);
    pack_a_kernel<<<dim3(HH / 16, LL / 32), 32, 0, stream>>>(st_w, awp_s);
    pack_a_kernel<<<dim3(HH / 16, LL / 32), 32, 0, stream>>>(tt_w, awp_t);

    stats_kernel<<<BB, CC, 0, stream>>>(x, mean, stdv);
    scan_kernel<<<BB, CC, 0, stream>>>(x, alpha, rev_w, rev_b, mean, stdv, seas, tren);

    const int nblocks = (int)(NBL / 16 / 2);
    mlp_kernel<<<nblocks, 64, 0, stream>>>(seas, ewp_s, se_b, pwp_s, sp_b, gp_s);
    mlp_kernel<<<nblocks, 64, 0, stream>>>(tren, ewp_t, te_b, pwp_t, tp_b, gp_t);

    dim3 tg(HH / 16, BB);
    temporal_kernel<<<tg, 128, 0, stream>>>(gp_s, gp_t, awp_s, awp_t, st_b, tt_b,
                                            rev_w, rev_b, mean, stdv, out);
}
